// TargetSequenceEncoder_73778948210981
// MI455X (gfx1250) — compile-verified, matrix operand order checked by reading, not run on hardware
//
#include <hip/hip_runtime.h>
#include <hip/hip_bf16.h>

#define BATCH 256
#define TT    128
#define EDIM  256
#define UNITS 1024
#define GATES 4096
#define KTOT  1280   // EDIM + UNITS

typedef __attribute__((ext_vector_type(16))) __bf16 v16bf;
typedef __attribute__((ext_vector_type(8)))  float  v8f;

union Frag {
    v16bf v;
    uint4 q[2];
};

__device__ __forceinline__ unsigned short f2bf_bits(float f) {
    union { float f; unsigned int u; } v;
    v.f = f;
    unsigned int r = v.u + 0x7FFFu + ((v.u >> 16) & 1u);   // round-to-nearest-even
    return (unsigned short)(r >> 16);
}

__device__ __forceinline__ float sigm(float x) {
    return 1.0f / (1.0f + __expf(-x));
}

// ---------------- preprocessing kernels ----------------

__global__ void zero_kernel(float* __restrict__ p, int n) {
    int i = blockIdx.x * blockDim.x + threadIdx.x;
    if (i < n) p[i] = 0.0f;
}

__global__ void lengths_kernel(const int* __restrict__ seq, int* __restrict__ lengths) {
    int b = threadIdx.x;            // 256 threads, one per batch row
    int cnt = 0;
    for (int t = 0; t < TT; ++t) cnt += (seq[b * TT + t] != 0);
    lengths[b] = cnt;
}

// WcatT[g][k] = bf16( k < E ? W[k][g] : U[k-E][g] ), K-contiguous per column g
__global__ void pack_wcat_kernel(const float* __restrict__ W,
                                 const float* __restrict__ U,
                                 unsigned short* __restrict__ WcatT) {
    long long idx = (long long)blockIdx.x * blockDim.x + threadIdx.x;
    if (idx >= (long long)GATES * KTOT) return;
    int g = (int)(idx / KTOT);
    int k = (int)(idx % KTOT);
    float v = (k < EDIM) ? W[(long long)k * GATES + g]
                         : U[(long long)(k - EDIM) * GATES + g];
    WcatT[idx] = f2bf_bits(v);
}

// fcWT[o][k] = bf16( fc_W[k][o] )
__global__ void pack_fcw_kernel(const float* __restrict__ fcW,
                                unsigned short* __restrict__ fcWT) {
    long long idx = (long long)blockIdx.x * blockDim.x + threadIdx.x;
    if (idx >= (long long)UNITS * UNITS) return;
    int o = (int)(idx / UNITS);
    int k = (int)(idx % UNITS);
    fcWT[idx] = f2bf_bits(fcW[(long long)k * UNITS + o]);
}

// ---------------- LSTM step ----------------
// grid = (16 b-tiles, 16 u-tiles), block = 256 threads (8 waves).
// Computes z = [x_t | h] @ [W;U] for a 16x64 (b,u) tile across all 4 gates,
// then the fused masked cell update.
__global__ __launch_bounds__(256)
void lstm_step_kernel(int t,
                      const int*            __restrict__ seq,
                      const float*          __restrict__ emb,
                      const float*          __restrict__ bias,
                      const unsigned short* __restrict__ WcatT,
                      const float*          __restrict__ h_in,
                      float*                __restrict__ h_out,
                      float*                __restrict__ c,
                      const int*            __restrict__ lengths,
                      float*                __restrict__ hseq /* (B,T,UNITS) in d_out */) {
    __shared__ unsigned short A_lds[16 * KTOT];      // 40 KB: [emb_row | h_row] bf16
    __shared__ float z_lds[4][16][64];               // 16 KB: gate pre-activations

    const int tid = threadIdx.x;
    const int bx  = blockIdx.x;       // rows  bx*16 .. +15
    const int by  = blockIdx.y;       // units by*64 .. +63

    // ---- stage A tile: 16 rows x 1280 K (embedding gather + h) ----
    {
        const int r  = tid >> 4;                 // 0..15 row
        const int kk = tid & 15;
        const int b  = bx * 16 + r;
        const int sidx = seq[b * TT + t];
        const float* embrow = emb + (long long)sidx * EDIM;
        const float* hrow   = h_in + (long long)b * UNITS;
        unsigned short* dst = &A_lds[r * KTOT];
        for (int k = kk; k < EDIM; k += 16)  dst[k]        = f2bf_bits(embrow[k]);
        for (int k = kk; k < UNITS; k += 16) dst[EDIM + k] = f2bf_bits(hrow[k]);
    }
    __syncthreads();

    const int w     = tid >> 5;        // wave 0..7
    const int lane  = tid & 31;
    const int gate0 = (w >> 2) * 2;    // waves 0-3 -> gates 0,1 ; waves 4-7 -> gates 2,3
    const int gate1 = gate0 + 1;
    const int usub  = w & 3;           // 16-wide u sub-tile
    const int n     = lane & 15;
    const int khalf = (lane >> 4) * 8;

    const long long col0 = (long long)(gate0 * UNITS + by * 64 + usub * 16 + n) * KTOT;
    const long long col1 = (long long)(gate1 * UNITS + by * 64 + usub * 16 + n) * KTOT;
    const unsigned short* arow = &A_lds[(lane & 15) * KTOT + khalf];

    v8f acc0 = {};
    v8f acc1 = {};

    for (int kc = 0; kc < KTOT; kc += 32) {
        Frag a, b0, b1;
        a.q[0] = *(const uint4*)(arow + kc);
        a.q[1] = *(const uint4*)(arow + kc + 16);
        const unsigned short* p0 = WcatT + col0 + kc + khalf;
        b0.q[0] = *(const uint4*)(p0);
        b0.q[1] = *(const uint4*)(p0 + 16);
        const unsigned short* p1 = WcatT + col1 + kc + khalf;
        b1.q[0] = *(const uint4*)(p1);
        b1.q[1] = *(const uint4*)(p1 + 16);
        acc0 = __builtin_amdgcn_wmma_f32_16x16x32_bf16(false, a.v, false, b0.v,
                                                       (short)0, acc0, false, false);
        acc1 = __builtin_amdgcn_wmma_f32_16x16x32_bf16(false, a.v, false, b1.v,
                                                       (short)0, acc1, false, false);
    }

    // spill z tiles to LDS so each thread can fuse all four gates
    {
        const int rbase = (lane >> 4) * 8;
        const int cc    = usub * 16 + n;
        #pragma unroll
        for (int j = 0; j < 8; ++j) {
            z_lds[gate0][rbase + j][cc] = acc0[j];
            z_lds[gate1][rbase + j][cc] = acc1[j];
        }
    }
    __syncthreads();

    // fused masked LSTM cell update
    for (int i = tid; i < 16 * 64; i += 256) {
        const int r  = i >> 6;
        const int uc = i & 63;
        const int b  = bx * 16 + r;
        const int u  = by * 64 + uc;
        const float zi = z_lds[0][r][uc] + bias[0 * UNITS + u];
        const float zf = z_lds[1][r][uc] + bias[1 * UNITS + u];
        const float zg = z_lds[2][r][uc] + bias[2 * UNITS + u];
        const float zo = z_lds[3][r][uc] + bias[3 * UNITS + u];
        const float ig = sigm(zi);
        const float fg = sigm(zf);
        const float gg = tanhf(zg);
        const float og = sigm(zo);
        const long long off = (long long)b * UNITS + u;
        const float cold = c[off];
        const float hold = h_in[off];
        const float cnew = fg * cold + ig * gg;
        const float hnew = og * tanhf(cnew);
        const bool valid = t < lengths[b];
        c[off]     = valid ? cnew : cold;
        h_out[off] = valid ? hnew : hold;
        hseq[((long long)b * TT + t) * UNITS + u] = valid ? hnew : 0.0f;
    }
}

// ---------------- final FC, in-place on d_out ----------------
// Each block owns 16 full rows (stages them to LDS before overwriting).
__global__ __launch_bounds__(256)
void fc_kernel(const unsigned short* __restrict__ fcWT,
               const float*          __restrict__ fcb,
               float*                __restrict__ out /* 32768 x 1024 */) {
    __shared__ unsigned short A_lds[16 * UNITS];     // 32 KB

    const int tid = threadIdx.x;
    const long long row0 = (long long)blockIdx.x * 16;

    {   // stage + convert the 16 input rows
        const int r  = tid >> 4;
        const int kk = tid & 15;
        const float* src = out + (row0 + r) * UNITS;
        unsigned short* dst = &A_lds[r * UNITS];
        for (int k = kk; k < UNITS; k += 16) dst[k] = f2bf_bits(src[k]);
    }
    __syncthreads();

    const int w     = tid >> 5;
    const int lane  = tid & 31;
    const int n     = lane & 15;
    const int khalf = (lane >> 4) * 8;
    const unsigned short* arow = &A_lds[(lane & 15) * UNITS + khalf];

    v8f acc[8];
    #pragma unroll
    for (int j = 0; j < 8; ++j) acc[j] = (v8f){};

    for (int kc = 0; kc < UNITS; kc += 32) {
        Frag a;
        a.q[0] = *(const uint4*)(arow + kc);
        a.q[1] = *(const uint4*)(arow + kc + 16);
        #pragma unroll
        for (int j = 0; j < 8; ++j) {
            const int col = w * 128 + j * 16 + n;
            const unsigned short* p = fcWT + (long long)col * UNITS + kc + khalf;
            Frag bf;
            bf.q[0] = *(const uint4*)(p);
            bf.q[1] = *(const uint4*)(p + 16);
            acc[j] = __builtin_amdgcn_wmma_f32_16x16x32_bf16(false, a.v, false, bf.v,
                                                             (short)0, acc[j], false, false);
        }
    }

    const int rbase = (lane >> 4) * 8;
    #pragma unroll
    for (int j = 0; j < 8; ++j) {
        const int col = w * 128 + j * 16 + n;
        const float bb = fcb[col];
        #pragma unroll
        for (int q = 0; q < 8; ++q) {
            out[(row0 + rbase + q) * UNITS + col] = acc[j][q] + bb;
        }
    }
}

// ---------------- launch ----------------
extern "C" void kernel_launch(void* const* d_in, const int* in_sizes, int n_in,
                              void* d_out, int out_size, void* d_ws, size_t ws_size,
                              hipStream_t stream) {
    (void)in_sizes; (void)n_in; (void)out_size; (void)ws_size;

    const int*   seq  = (const int*)  d_in[0];
    const float* emb  = (const float*)d_in[1];
    const float* W    = (const float*)d_in[2];
    const float* U    = (const float*)d_in[3];
    const float* bias = (const float*)d_in[4];
    const float* fcW  = (const float*)d_in[5];
    const float* fcb  = (const float*)d_in[6];
    float* out = (float*)d_out;

    // workspace layout (256B aligned chunks)
    char* ws = (char*)d_ws;
    int*   lengths = (int*)ws;                                 ws += 4096;
    float* h0      = (float*)ws;                               ws += (size_t)BATCH * UNITS * 4;  // 1 MB
    float* h1      = (float*)ws;                               ws += (size_t)BATCH * UNITS * 4;  // 1 MB
    float* cbuf    = (float*)ws;                               ws += (size_t)BATCH * UNITS * 4;  // 1 MB
    unsigned short* WcatT = (unsigned short*)ws;               ws += (size_t)GATES * KTOT * 2;   // 10 MB
    unsigned short* fcWT  = (unsigned short*)ws;               // 2 MB

    // h0, h1, c are contiguous -> one zero pass
    {
        const int nz = 3 * BATCH * UNITS;
        zero_kernel<<<(nz + 255) / 256, 256, 0, stream>>>(h0, nz);
    }
    lengths_kernel<<<1, 256, 0, stream>>>(seq, lengths);
    {
        const long long n = (long long)GATES * KTOT;
        pack_wcat_kernel<<<(int)((n + 255) / 256), 256, 0, stream>>>(W, U, WcatT);
    }
    {
        const long long n = (long long)UNITS * UNITS;
        pack_fcw_kernel<<<(int)((n + 255) / 256), 256, 0, stream>>>(fcW, fcWT);
    }

    dim3 sgrid(BATCH / 16, UNITS / 64);   // 16 x 16 blocks
    for (int t = 0; t < TT; ++t) {
        const float* hin  = (t & 1) ? h1 : h0;
        float*       hout = (t & 1) ? h0 : h1;
        lstm_step_kernel<<<sgrid, 256, 0, stream>>>(t, seq, emb, bias, WcatT,
                                                    hin, hout, cbuf, lengths, out);
    }

    fc_kernel<<<(BATCH * TT) / 16, 256, 0, stream>>>(fcWT, fcb, out);
}
